// Attention_82016695484963
// MI455X (gfx1250) — matrix operand order repaired and checked by reading, not compiled, not run on hardware
//
#include <hip/hip_runtime.h>

// ---------------------------------------------------------------------------
// Types for CDNA5 WMMA / TDM
// ---------------------------------------------------------------------------
typedef __attribute__((ext_vector_type(4)))  unsigned int v4u;
typedef __attribute__((ext_vector_type(8)))  unsigned int v8u;
typedef __attribute__((ext_vector_type(16))) __bf16       v16b;
typedef __attribute__((ext_vector_type(8)))  float        v8f;
typedef __attribute__((ext_vector_type(4)))  unsigned int u32x4;
typedef __attribute__((ext_vector_type(8)))  int          i32x8;
typedef __attribute__((ext_vector_type(4)))  int          i32x4;

// Problem constants
#define Bb 4
#define Tt 2048
#define Cc 1024
#define Hh 16
#define Dd 64

__device__ __forceinline__ unsigned short f2bf(float f) {
    unsigned int u = __builtin_bit_cast(unsigned int, f);
    unsigned int r = (u + 0x7FFFu + ((u >> 16) & 1u)) >> 16;  // RNE
    return (unsigned short)r;
}

__device__ __forceinline__ v16b combine16(v4u lo, v4u hi) {
    v8u r;
    r[0] = lo[0]; r[1] = lo[1]; r[2] = lo[2]; r[3] = lo[3];
    r[4] = hi[0]; r[5] = hi[1]; r[6] = hi[2]; r[7] = hi[3];
    return __builtin_bit_cast(v16b, r);
}

// Escape an LDS pointer so alias analysis must assume external writers (TDM).
__device__ __forceinline__ void lds_escape(void* p) {
    asm volatile("" : "+v"(p) :: "memory");
}
// Compile-time memory barrier: "LDS contents changed here" (TDM completion).
__device__ __forceinline__ void mem_clobber() {
    asm volatile("" ::: "memory");
}

// A-fragment: 16x32 bf16, row-major source, stride in elements.
// Lane half h: row = ln (0..15), K-chunks {kbase+8h .. +7} and {+16..+23}.
__device__ __forceinline__ v16b load_frag_a(const unsigned short* base, int stride,
                                            int ln, int half, int kbase) {
    const unsigned short* row = base + (long)ln * stride;
    int c0 = kbase + half * 8;
    v4u lo = *(const v4u*)(row + c0);
    v4u hi = *(const v4u*)(row + c0 + 16);
    return combine16(lo, hi);
}

// B-fragment: 32x16 bf16 where column n is stored contiguously (row n of Bt).
// Lane half h holds K-chunks {kbase+8h .. +7} and {+16..+23} (matches A).
__device__ __forceinline__ v16b load_frag_b(const unsigned short* base, int stride,
                                            int ln, int half, int kbase) {
    const unsigned short* row = base + (long)ln * stride;
    int c0 = kbase + half * 8;
    v4u lo = *(const v4u*)(row + c0);
    v4u hi = *(const v4u*)(row + c0 + 16);
    return combine16(lo, hi);
}

__device__ __forceinline__ v8f wmma_bf16(v16b a, v16b b, v8f c) {
    return __builtin_amdgcn_wmma_f32_16x16x32_bf16(
        /*neg_a=*/false, a, /*neg_b=*/false, b,
        /*c_mod=*/(short)0, c, /*reuse_a=*/false, /*reuse_b=*/false);
}

// ---------------------------------------------------------------------------
// TDM: issue a 2-D bf16 tile load (tile_d0 contiguous elems x tile_d1 rows)
// from global (row stride stride0 elements) into LDS byte offset lds_off.
// D# layout per CDNA5 ISA. Tracked by TENSORcnt.
// This toolchain's builtin is the 6-arg form:
//   (u32x4 g0, i32x8 g1, i32x4 g2, i32x4 g3, i32x8 extra, i32 cpol)
// ---------------------------------------------------------------------------
__device__ __forceinline__ void tdm_load_2d(unsigned lds_off, const void* gptr,
                                            unsigned tensor_d0, unsigned tensor_d1,
                                            unsigned stride0,
                                            unsigned tile_d0, unsigned tile_d1) {
    unsigned long long ga = (unsigned long long)(uintptr_t)gptr;
    u32x4 g0;
    g0[0] = 1u;                                  // count=1, user mode, no gather
    g0[1] = lds_off;                             // lds_addr (bytes)
    g0[2] = (unsigned)ga;                        // global_addr[31:0]
    g0[3] = (unsigned)(ga >> 32) | (2u << 30);   // global_addr[56:32] | type=2
    i32x8 g1;
    g1[0] = 0x10000;                                           // data_size=2B
    g1[1] = (int)((tensor_d0 & 0xFFFFu) << 16);                // tensor_dim0 lo
    g1[2] = (int)(((tensor_d0 >> 16) & 0xFFFFu) |
                  ((tensor_d1 & 0xFFFFu) << 16));              // d0 hi | d1 lo
    g1[3] = (int)(((tensor_d1 >> 16) & 0xFFFFu) |
                  ((tile_d0 & 0xFFFFu) << 16));                // d1 hi | tile_dim0
    g1[4] = (int)(tile_d1 & 0xFFFFu);                          // tile_dim1, tile_dim2=0
    g1[5] = (int)stride0;                                      // tensor_dim0_stride lo
    g1[6] = 0;                                                 // stride0 hi | stride1 lo
    g1[7] = 0;
    i32x4 z4 = {};
    i32x8 z8 = {};
    __builtin_amdgcn_tensor_load_to_lds(g0, g1, z4, z4, z8, 0);
}

// ---------------------------------------------------------------------------
// Conversion kernels
// ---------------------------------------------------------------------------
__global__ void cvt_bf16_kernel(const float* __restrict__ in,
                                unsigned short* __restrict__ out, long n) {
    long i = (long)blockIdx.x * blockDim.x + threadIdx.x;
    if (i < n) out[i] = f2bf(in[i]);
}

// out(N,K) bf16 = transpose of in(K,N) fp32
__global__ void transpose_bf16_kernel(const float* __restrict__ in,
                                      unsigned short* __restrict__ out,
                                      int K, int N) {
    long i = (long)blockIdx.x * blockDim.x + threadIdx.x;
    if (i >= (long)K * N) return;
    int n = (int)(i / K);
    int k = (int)(i % K);
    out[i] = f2bf(in[(long)k * N + n]);
}

// ---------------------------------------------------------------------------
// Shared GEMM core: block tile 64(M) x 256(N), 8 waves (2x4), wave tile 32x64.
// A (64x32) and B (256x32) K-step tiles staged in LDS by TDM, double-buffered.
// LDS: A: [0,4K) [4K,8K) ; B: [8K,24K) [24K,40K)
// ---------------------------------------------------------------------------
#define LDS_A0   0u
#define LDS_A1   4096u
#define LDS_B0   8192u
#define LDS_B1   24576u

__device__ __forceinline__ void gemm_core(const unsigned short* __restrict__ A,
                                          const unsigned short* __restrict__ Bt,
                                          unsigned Ntot, unsigned short* lds,
                                          long mblock, long nblock,
                                          int w, int wm, int wn, int ln, int half,
                                          v8f acc[2][4]) {
    const int KSTEPS = Cc / 32;
    lds_escape(lds);   // TDM writes this buffer behind the compiler's back
    if (w == 0) {
        tdm_load_2d(LDS_A0, A + mblock * Cc, Cc, Bb * Tt, Cc, 32, 64);
        tdm_load_2d(LDS_B0, Bt + nblock * Cc, Cc, Ntot, Cc, 32, 256);
    }
    for (int s = 0; s < KSTEPS; ++s) {
        int buf = s & 1;
        if (w == 0) {
            if (s + 1 < KSTEPS) {
                long kk2 = (long)(s + 1) * 32;
                tdm_load_2d(buf ? LDS_A0 : LDS_A1, A + mblock * Cc + kk2,
                            Cc, Bb * Tt, Cc, 32, 64);
                tdm_load_2d(buf ? LDS_B0 : LDS_B1, Bt + nblock * Cc + kk2,
                            Cc, Ntot, Cc, 32, 256);
                __builtin_amdgcn_s_wait_tensorcnt(2);  // current-buffer pair done
            } else {
                __builtin_amdgcn_s_wait_tensorcnt(0);
            }
        }
        __syncthreads();
        mem_clobber();   // LDS tiles (re)filled by TDM: force fresh ds loads
        const unsigned short* la = lds + (buf ? (LDS_A1 / 2) : (LDS_A0 / 2));
        const unsigned short* lb = lds + (buf ? (LDS_B1 / 2) : (LDS_B0 / 2));
        v16b a0 = load_frag_a(la + (wm * 32) * 32,      32, ln, half, 0);
        v16b a1 = load_frag_a(la + (wm * 32 + 16) * 32, 32, ln, half, 0);
#pragma unroll
        for (int nt = 0; nt < 4; ++nt) {
            v16b bf = load_frag_b(lb + (wn * 64 + nt * 16) * 32, 32, ln, half, 0);
            acc[0][nt] = wmma_bf16(a0, bf, acc[0][nt]);
            acc[1][nt] = wmma_bf16(a1, bf, acc[1][nt]);
        }
        __syncthreads();
    }
}

// ---------------------------------------------------------------------------
// GEMM1: QKV = x(8192,1024) @ w_qkv + b, scattered into Q, K (B,H,T,D) and
// V^T (B,H,D,T), all bf16.
// ---------------------------------------------------------------------------
__global__ __launch_bounds__(256) void gemm_qkv_kernel(
    const unsigned short* __restrict__ A,    // 8192 x 1024 bf16
    const unsigned short* __restrict__ Bt,   // 3072 x 1024 bf16 (wT)
    const float* __restrict__ bias,          // 3072
    unsigned short* __restrict__ Qb,
    unsigned short* __restrict__ Kb,
    unsigned short* __restrict__ VTb) {
    __shared__ __align__(16) unsigned short lds[20480];  // 40 KB
    int lane = threadIdx.x & 31;
    int w    = threadIdx.x >> 5;
    int half = lane >> 4;
    int ln   = lane & 15;
    int wm = w & 1;
    int wn = w >> 1;
    long mblock = (long)blockIdx.y * 64;
    long nblock = (long)blockIdx.x * 256;
    long mbase = mblock + wm * 32;
    long nbase = nblock + wn * 64;

    v8f acc[2][4] = {};
    gemm_core(A, Bt, 3 * Cc, lds, mblock, nblock, w, wm, wn, ln, half, acc);

#pragma unroll
    for (int nt = 0; nt < 4; ++nt) {
        int n   = (int)nbase + nt * 16 + ln;
        float bv = bias[n];
        int sec = n >> 10;         // 0=Q 1=K 2=V
        int nc  = n & 1023;
        int h   = nc >> 6;
        int d   = nc & 63;
#pragma unroll
        for (int mt = 0; mt < 2; ++mt) {
#pragma unroll
            for (int r = 0; r < 8; ++r) {
                long m = mbase + mt * 16 + r + half * 8;
                int b = (int)(m >> 11);
                int t = (int)(m & 2047);
                unsigned short v = f2bf(acc[mt][nt][r] + bv);
                long bh = (long)b * Hh + h;
                if (sec == 0)      Qb[(bh * Tt + t) * Dd + d] = v;
                else if (sec == 1) Kb[(bh * Tt + t) * Dd + d] = v;
                else               VTb[(bh * Dd + d) * Tt + t] = v;
            }
        }
    }
}

// ---------------------------------------------------------------------------
// GEMM2: out(8192,1024) fp32 = attn(8192,1024)bf16 @ w_proj + b_proj
// ---------------------------------------------------------------------------
__global__ __launch_bounds__(256) void gemm_proj_kernel(
    const unsigned short* __restrict__ A,    // 8192 x 1024 bf16
    const unsigned short* __restrict__ Bt,   // 1024 x 1024 bf16 (w_projT)
    const float* __restrict__ bias,          // 1024
    float* __restrict__ out) {               // 8192 x 1024 fp32
    __shared__ __align__(16) unsigned short lds[20480];  // 40 KB
    int lane = threadIdx.x & 31;
    int w    = threadIdx.x >> 5;
    int half = lane >> 4;
    int ln   = lane & 15;
    int wm = w & 1;
    int wn = w >> 1;
    long mblock = (long)blockIdx.y * 64;
    long nblock = (long)blockIdx.x * 256;
    long mbase = mblock + wm * 32;
    long nbase = nblock + wn * 64;

    v8f acc[2][4] = {};
    gemm_core(A, Bt, Cc, lds, mblock, nblock, w, wm, wn, ln, half, acc);

#pragma unroll
    for (int nt = 0; nt < 4; ++nt) {
        int n = (int)nbase + nt * 16 + ln;
        float bv = bias[n];
#pragma unroll
        for (int mt = 0; mt < 2; ++mt) {
#pragma unroll
            for (int r = 0; r < 8; ++r) {
                long m = mbase + mt * 16 + r + half * 8;
                out[m * Cc + n] = acc[mt][nt][r] + bv;
            }
        }
    }
}

// ---------------------------------------------------------------------------
// Flash attention: one wave per 16-query tile per (b,h). 8 waves/block.
// ---------------------------------------------------------------------------
__global__ __launch_bounds__(256) void attn_kernel(
    const unsigned short* __restrict__ Q,    // (B*H, T, D) bf16
    const unsigned short* __restrict__ K,    // (B*H, T, D) bf16
    const unsigned short* __restrict__ VT,   // (B*H, D, T) bf16
    unsigned short* __restrict__ OUT) {      // (B, T, C)   bf16
    __shared__ __align__(16) unsigned short ldsP[8][16 * 32];

    int tid  = threadIdx.x;
    int lane = tid & 31;
    int wib  = tid >> 5;
    int half = lane >> 4;
    int ln   = lane & 15;
    int wid  = blockIdx.x * 8 + wib;
    int qt   = wid & 127;            // query tile within (b,h)
    int bh   = wid >> 7;             // 0..63

    const unsigned short* q  = Q  + (long)bh * Tt * Dd;
    const unsigned short* k  = K  + (long)bh * Tt * Dd;
    const unsigned short* vt = VT + (long)bh * Dd * Tt;

    // Q A-fragments (d 0..31 and 32..63), reused across all key blocks.
    v16b aq0 = load_frag_a(q + (long)qt * 16 * Dd, Dd, ln, half, 0);
    v16b aq1 = load_frag_a(q + (long)qt * 16 * Dd, Dd, ln, half, 32);

    v8f acc[4] = {};
    float ms[8], ls[8];
#pragma unroll
    for (int r = 0; r < 8; ++r) { ms[r] = -3.0e38f; ls[r] = 0.0f; }

    const float scale = 0.125f;      // 1/sqrt(64)
    int kbend = qt >> 1;             // last 32-key block touching the diagonal
    int qrow_base = qt * 16 + half * 8;

    for (int kb = 0; kb <= kbend; ++kb) {
        int keyb = kb * 32;
        // S = Q K^T for 32 keys: two 16-key tiles, contraction over d in 2 steps.
        v8f s0 = {}, s1 = {};
        s0 = wmma_bf16(aq0, load_frag_b(k + (long)keyb * Dd,        Dd, ln, half, 0),  s0);
        s0 = wmma_bf16(aq1, load_frag_b(k + (long)keyb * Dd,        Dd, ln, half, 32), s0);
        s1 = wmma_bf16(aq0, load_frag_b(k + (long)(keyb + 16) * Dd, Dd, ln, half, 0),  s1);
        s1 = wmma_bf16(aq1, load_frag_b(k + (long)(keyb + 16) * Dd, Dd, ln, half, 32), s1);

        bool lastblk = (kb == kbend);  // uniform per wave
        float p0[8], p1[8];
#pragma unroll
        for (int r = 0; r < 8; ++r) {
            float x0 = s0[r] * scale;
            float x1 = s1[r] * scale;
            if (lastblk) {
                int qrow = qrow_base + r;
                x0 = (keyb + ln       > qrow) ? -3.0e38f : x0;
                x1 = (keyb + 16 + ln  > qrow) ? -3.0e38f : x1;
            }
            float bm = fmaxf(x0, x1);
#pragma unroll
            for (int msk = 1; msk < 16; msk <<= 1)
                bm = fmaxf(bm, __shfl_xor(bm, msk, 16));
            float mn = fmaxf(ms[r], bm);
            float e0 = __expf(x0 - mn);
            float e1 = __expf(x1 - mn);
            float rs = e0 + e1;
#pragma unroll
            for (int msk = 1; msk < 16; msk <<= 1)
                rs += __shfl_xor(rs, msk, 16);
            float al = __expf(ms[r] - mn);
            ls[r] = ls[r] * al + rs;
            ms[r] = mn;
            p0[r] = e0; p1[r] = e1;
#pragma unroll
            for (int j = 0; j < 4; ++j) acc[j][r] *= al;
        }

        // C-layout -> A-fragment relayout through wave-private LDS.
        unsigned short* lp = ldsP[wib];
#pragma unroll
        for (int r = 0; r < 8; ++r) {
            int row = r + half * 8;
            lp[row * 32 + ln]      = f2bf(p0[r]);
            lp[row * 32 + 16 + ln] = f2bf(p1[r]);
        }
        v16b pf = load_frag_a(lp, 32, ln, half, 0);

        // O += P (16x32) x V (32 keys x 64 d); V columns contiguous in V^T rows.
#pragma unroll
        for (int j = 0; j < 4; ++j) {
            v16b bv = load_frag_b(vt + (long)(j * 16) * Tt, Tt, ln, half, keyb);
            acc[j] = wmma_bf16(pf, bv, acc[j]);
        }
    }

    // Normalize and store to (B, T, C) bf16 for the projection GEMM.
    int b = bh >> 4, h = bh & 15;
#pragma unroll
    for (int j = 0; j < 4; ++j) {
#pragma unroll
        for (int r = 0; r < 8; ++r) {
            int row = r + half * 8;
            int t = qt * 16 + row;
            int d = j * 16 + ln;
            float o = acc[j][r] / ls[r];
            OUT[((long)b * Tt + t) * Cc + h * Dd + d] = f2bf(o);
        }
    }
}

// ---------------------------------------------------------------------------
// Host launcher
// ---------------------------------------------------------------------------
extern "C" void kernel_launch(void* const* d_in, const int* in_sizes, int n_in,
                              void* d_out, int out_size, void* d_ws, size_t ws_size,
                              hipStream_t stream) {
    const float* x      = (const float*)d_in[0];  // (B,T,C)
    const float* w_qkv  = (const float*)d_in[1];  // (C,3C)
    const float* b_qkv  = (const float*)d_in[2];  // (3C)
    const float* w_proj = (const float*)d_in[3];  // (C,C)
    const float* b_proj = (const float*)d_in[4];  // (C)
    float* out = (float*)d_out;

    char* ws = (char*)d_ws;
    const long NTOK = (long)Bb * Tt;              // 8192
    unsigned short* xb     = (unsigned short*)(ws);                     // 16 MB
    unsigned short* wqkvT  = (unsigned short*)(ws + 16777216);          //  6 MB
    unsigned short* wprojT = (unsigned short*)(ws + 23068672);          //  2 MB
    unsigned short* Qb     = (unsigned short*)(ws + 25165824);          // 16 MB
    unsigned short* Kb     = (unsigned short*)(ws + 41943040);          // 16 MB
    unsigned short* VTb    = (unsigned short*)(ws + 58720256);          // 16 MB
    unsigned short* attnb  = (unsigned short*)(ws + 75497472);          // 16 MB

    // 1) fp32 -> bf16 conversions / weight transposes
    long nx = NTOK * Cc;                                   // 8388608
    cvt_bf16_kernel<<<(unsigned)((nx + 255) / 256), 256, 0, stream>>>(x, xb, nx);
    long nwq = (long)Cc * 3 * Cc;                          // 3145728
    transpose_bf16_kernel<<<(unsigned)((nwq + 255) / 256), 256, 0, stream>>>(
        w_qkv, wqkvT, Cc, 3 * Cc);
    long nwp = (long)Cc * Cc;                              // 1048576
    transpose_bf16_kernel<<<(unsigned)((nwp + 255) / 256), 256, 0, stream>>>(
        w_proj, wprojT, Cc, Cc);

    // 2) QKV projection (TDM-staged tiles, scatter to Q, K, V^T)
    gemm_qkv_kernel<<<dim3(12, 128), 256, 0, stream>>>(xb, wqkvT, b_qkv, Qb, Kb, VTb);

    // 3) Flash attention: B*H*(T/16) = 8192 wave-tiles, 8 waves/block
    attn_kernel<<<1024, 256, 0, stream>>>(Qb, Kb, VTb, attnb);

    // 4) Output projection + bias -> fp32
    gemm_proj_kernel<<<dim3(4, 128), 256, 0, stream>>>(attnb, wprojT, b_proj, out);
}
